// FConv2d_72919954752251
// MI455X (gfx1250) — matrix operand order repaired and checked by reading, not compiled, not run on hardware
//
#include <hip/hip_runtime.h>

// FFT-conv rewritten as 3D circular convolution mapped onto v_wmma_f32_16x16x32_f16.
// out[b, n*32+j, h, w] = sum_{c'<16, kh<9, kw<9}
//     weight[n,c',kh,kw] * x[b,(4j-c')%128,(h-kh)%32,(w-kw)%32]
//
// Grouping 4 consecutive j -> 32 output rows sharing a 28-channel (padded to 32)
// circular x-window => M=32 (two 16-row WMMA tiles), K = 81 taps x 32 channels,
// N = 1024 spatial positions per (batch, group), split in 2 for occupancy.

typedef __attribute__((ext_vector_type(16))) _Float16 v16h;
typedef __attribute__((ext_vector_type(8)))  float    v8f;

#define LDIM   32
#define CIN    128
#define KS     9
#define TAPS   81
#define GROUPS 8     // groups of 4 j-values
#define UPAD   32    // padded window channels (28 real)

// ---------------------------------------------------------------------------
// Prep: expand weights into WMMA A-fragment-major layout.
// wexp[((g*2+rt)*81 + t)*32 + lane] is a v16h (32B) holding lane's A fragment
// for row-tile rt of group g, K-chunk t. A layout used here (contiguous K, same as B):
//   lane<16 : row = lane,    element i -> K = i
//   lane>=16: row = lane-16, element i -> K = i + 16
// K index == padded window channel u; c' = 15 + 4d - u (zero outside [0,16)).
// ---------------------------------------------------------------------------
__global__ __launch_bounds__(256) void expand_weight_kernel(
    const float* __restrict__ w, _Float16* __restrict__ wexp) {
    int idx = blockIdx.x * blockDim.x + threadIdx.x;      // (g, rt, t, lane)
    if (idx >= GROUPS * 2 * TAPS * 32) return;
    int lane = idx & 31;
    int q    = idx >> 5;
    int t    = q % TAPS;
    int q2   = q / TAPS;
    int rt   = q2 & 1;
    int kh = t / KS, kw = t % KS;
    int half = lane >> 4;
    int r = rt * 16 + (lane & 15);   // row within the 32-row group: r = d*8 + n
    int d = r >> 3;
    int n = r & 7;
    v16h frag;
    #pragma unroll
    for (int i = 0; i < 16; ++i) {
        int u = i + (half << 4);  // K index of element i
        int c = 15 + 4 * d - u;                           // original c'
        float val = (c >= 0 && c < 16) ? w[((n * 16 + c) * KS + kh) * KS + kw] : 0.0f;
        frag[i] = (_Float16)val;
    }
    *reinterpret_cast<v16h*>(wexp + (size_t)idx * 16) = frag;
}

// ---------------------------------------------------------------------------
// Main conv. Block = (group g, batch b, spatial half z), 256 threads = 8 wave32.
// LDS: x tile as xs[pos][u] f16 (64KB) -> B fragment = contiguous 32B per lane.
// ---------------------------------------------------------------------------
__global__ __launch_bounds__(256) void fconv_wmma_kernel(
    const float* __restrict__ x, const _Float16* __restrict__ wexp,
    float* __restrict__ out) {
    __shared__ __align__(32) _Float16 xs[LDIM * LDIM * UPAD];  // 65536 B

    const int g   = blockIdx.x;
    const int b   = blockIdx.y;
    const int z   = blockIdx.z;         // spatial half: positions [z*512, z*512+512)
    const int tid = threadIdx.x;
    const int wave = tid >> 5;          // wave32
    const int lane = tid & 31;

    // ---- Stage x window into LDS (fp32 -> f16, channel-innermost layout). ----
    // lane = window channel u, loop = position: each ds-store instruction writes
    // a contiguous 64B run (conflict-free); each lane streams float4 from its
    // own channel plane (L2-resident, 128KB/block total).
    {
        const float* xb = x + (size_t)b * CIN * (LDIM * LDIM);
        int cc = (16 * g - 15 + lane + 128) & 127;        // circular channel window
        const float* xc = xb + cc * (LDIM * LDIM);
        const int pbase = wave * 128;                     // 128 positions per wave
        #pragma unroll 4
        for (int p4 = 0; p4 < 128; p4 += 4) {
            const float4 v = *reinterpret_cast<const float4*>(xc + pbase + p4);
            xs[(pbase + p4 + 0) * UPAD + lane] = (_Float16)v.x;
            xs[(pbase + p4 + 1) * UPAD + lane] = (_Float16)v.y;
            xs[(pbase + p4 + 2) * UPAD + lane] = (_Float16)v.z;
            xs[(pbase + p4 + 3) * UPAD + lane] = (_Float16)v.w;
        }
    }
    __syncthreads();

    const int half = lane >> 4;
    const int l15  = lane & 15;
    const int colbase = z * 512 + wave * 64;  // 4 col-tiles of 16 positions per wave

    int hv[4], wv[4];
    #pragma unroll
    for (int ct = 0; ct < 4; ++ct) {
        int p = colbase + ct * 16 + l15;
        hv[ct] = p >> 5;
        wv[ct] = p & 31;
    }

    v8f zero = {};
    v8f acc[2][4];
    #pragma unroll
    for (int rt = 0; rt < 2; ++rt)
        #pragma unroll
        for (int ct = 0; ct < 4; ++ct)
            acc[rt][ct] = zero;

    const v16h* __restrict__ Afrag = reinterpret_cast<const v16h*>(wexp);
    const size_t abase = (size_t)(g * 2) * TAPS * 32 + lane;

    int kh = 0, kw = 0;
    for (int t = 0; t < TAPS; ++t) {
        // A fragments for both 16-row tiles (32B contiguous per lane, L1-cached)
        v16h a0 = Afrag[abase + (size_t)t * 32];
        v16h a1 = Afrag[abase + (size_t)(TAPS + t) * 32];
        #pragma unroll
        for (int ct = 0; ct < 4; ++ct) {
            int hh = (hv[ct] - kh) & 31;              // circular spatial shift
            int ww = (wv[ct] - kw) & 31;
            // B fragment: col = l15, K = u = half*16 + i -> contiguous 32B in LDS;
            // a full wave reads one contiguous 1KB block -> conflict-free b128s.
            const v16h bfrag = *reinterpret_cast<const v16h*>(
                &xs[(hh * 32 + ww) * UPAD + half * 16]);
            acc[0][ct] = __builtin_amdgcn_wmma_f32_16x16x32_f16(
                false, a0, false, bfrag, (short)0, acc[0][ct], false, false);
            acc[1][ct] = __builtin_amdgcn_wmma_f32_16x16x32_f16(
                false, a1, false, bfrag, (short)0, acc[1][ct], false, false);
        }
        if (++kw == KS) { kw = 0; ++kh; }
    }

    // ---- Write out. C/D layout: VGPR v, lane -> (M = v + 8*half, N = l15) ----
    float* ob = out + (size_t)b * 256 * (LDIM * LDIM);
    #pragma unroll
    for (int rt = 0; rt < 2; ++rt) {
        #pragma unroll
        for (int v = 0; v < 8; ++v) {
            int r  = rt * 16 + v + 8 * half;          // row within group
            int oc = (r & 7) * 32 + 4 * g + (r >> 3); // n*32 + j
            float* orow = ob + (size_t)oc * (LDIM * LDIM);
            #pragma unroll
            for (int ct = 0; ct < 4; ++ct) {
                int p = colbase + ct * 16 + l15;      // coalesced 64B per half-wave
                orow[p] = acc[rt][ct][v];
            }
        }
    }
}

extern "C" void kernel_launch(void* const* d_in, const int* in_sizes, int n_in,
                              void* d_out, int out_size, void* d_ws, size_t ws_size,
                              hipStream_t stream) {
    const float* x = (const float*)d_in[0];      // (32,128,32,32) fp32
    const float* w = (const float*)d_in[1];      // (8,16,9,9) fp32
    float* out = (float*)d_out;                  // (32,256,32,32) fp32
    _Float16* wexp = (_Float16*)d_ws;            // 1.3 MB expanded weights

    // 8 groups * 2 row-tiles * 81 taps * 32 lanes = 41472 fragments
    expand_weight_kernel<<<162, 256, 0, stream>>>(w, wexp);
    // grid: (group, batch, spatial-half); block: 8 wave32
    fconv_wmma_kernel<<<dim3(GROUPS, 32, 2), 256, 0, stream>>>(x, wexp, out);
}
